// Net_25993142075982
// MI455X (gfx1250) — compile-verified, matrix operand order checked by reading, not run on hardware
//
#include <hip/hip_runtime.h>
#include <hip/hip_bf16.h>
#include <stdint.h>

// ---------------------------------------------------------------------------
// Types / helpers
// ---------------------------------------------------------------------------
typedef __attribute__((ext_vector_type(16))) __bf16    v16bf;
typedef __attribute__((ext_vector_type(8)))  __bf16    v8bf;
typedef __attribute__((ext_vector_type(16))) uint16_t  v16u;
typedef __attribute__((ext_vector_type(8)))  float     v8f;

#define DEV __device__ __forceinline__

DEV uint16_t f2bf(float f) {
    union { float f; uint32_t u; } v; v.f = f;
    uint32_t u = v.u;
    uint32_t r = u + 0x7fffu + ((u >> 16) & 1u);   // round-to-nearest-even
    return (uint16_t)(r >> 16);
}
DEV float bf2f(uint16_t h) {
    union { uint32_t u; float f; } v; v.u = ((uint32_t)h) << 16; return v.f;
}

// K index of fragment element j (0..15) for 16x16x32 bf16 WMMA A/B operands.
DEV int kmap(int j, int lane) {
    return ((j & 8) << 1) | ((lane & 16) >> 1) | (j & 7);
}

DEV v8f wmma_bf16(v16u a, v16u b, v8f c) {
    union { v16u u; v16bf bf; } A, B;
    A.u = a; B.u = b;
    return __builtin_amdgcn_wmma_f32_16x16x32_bf16(
        false, A.bf, false, B.bf, (short)0, c, false, false);
}

// ---------------------------------------------------------------------------
// Elementwise prep kernels
// ---------------------------------------------------------------------------
__global__ void ea_sum_kernel(const float* __restrict__ a, const float* __restrict__ b,
                              float* __restrict__ o, int n) {
    int t = blockIdx.x * blockDim.x + threadIdx.x;
    if (t < n) o[t] = a[t] + b[t];
}

__global__ void cvt_bf16_kernel(const float* __restrict__ s, uint16_t* __restrict__ d, int n) {
    int t = blockIdx.x * blockDim.x + threadIdx.x;
    if (t < n) d[t] = f2bf(s[t]);
}

// Swizzle a row-major f32 [K,N] matrix into WMMA B-fragment order (bf16).
__global__ void swz_w_kernel(const float* __restrict__ B, uint16_t* __restrict__ o,
                             int K, int N) {
    int t = blockIdx.x * blockDim.x + threadIdx.x;
    if (t >= K * N) return;
    int tile = t >> 9, r = t & 511, lane = r >> 4, j = r & 15;
    int tilesN = N >> 4;
    int kt = tile / tilesN, tn = tile - kt * tilesN;
    int kk = (kt << 5) + kmap(j, lane);
    int n  = (tn << 4) + (lane & 15);
    o[t] = f2bf(B[(size_t)kk * N + n]);
}

// Swizzled build of [KP, DOUT] = [reshape(w3,[64*DIN,DOUT]) ; b3 ; zero-pad]
__global__ void w3cat_swz_kernel(const float* __restrict__ w3, const float* __restrict__ b3,
                                 uint16_t* __restrict__ o, int DIN, int DOUT, int KP) {
    int t = blockIdx.x * blockDim.x + threadIdx.x;
    if (t >= KP * DOUT) return;
    int tile = t >> 9, r = t & 511, lane = r >> 4, j = r & 15;
    int tilesN = DOUT >> 4;
    int kt = tile / tilesN, tn = tile - kt * tilesN;
    int kk = (kt << 5) + kmap(j, lane);
    int n  = (tn << 4) + (lane & 15);
    float v = 0.f;
    if (kk < 64 * DIN)      v = w3[(size_t)kk * DOUT + n];
    else if (kk < 65 * DIN) v = b3[(size_t)(kk - 64 * DIN) * DOUT + n];
    o[t] = f2bf(v);
}

// Edge MLP layer 1: h1 = relu(ea @ w1 + b1), K=3 (too small for WMMA)
__global__ void edge_h1_kernel(const float* __restrict__ ea, const float* __restrict__ w1,
                               const float* __restrict__ b1, uint16_t* __restrict__ h1) {
    int t = blockIdx.x * blockDim.x + threadIdx.x;
    if (t >= 8192 * 64) return;
    int e = t >> 6, c = t & 63;
    float acc = b1[c] + ea[e*3+0]*w1[c] + ea[e*3+1]*w1[64+c] + ea[e*3+2]*w1[128+c];
    h1[t] = f2bf(acc > 0.f ? acc : 0.f);
}

// ---------------------------------------------------------------------------
// Generic WMMA GEMM: two adjacent N-tiles per wave (independent accumulators
// hide the WMMA->WMMA RAW hazard). A row-major bf16, B pre-swizzled bf16.
// ---------------------------------------------------------------------------
__global__ __launch_bounds__(256)
void wmma_gemm_kernel(const uint16_t* __restrict__ A, const uint16_t* __restrict__ Bs,
                      const float* __restrict__ bias,
                      uint16_t* __restrict__ Cb, float* __restrict__ Cf,
                      int M, int N, int K, int relu) {
    int lane   = threadIdx.x & 31;
    int wave   = (blockIdx.x * blockDim.x + threadIdx.x) >> 5;
    int nwaves = (gridDim.x * blockDim.x) >> 5;
    int tilesN = N >> 4;
    int pairsN = tilesN >> 1;
    int pairs  = (M >> 4) * pairsN;
    int row = lane & 15, off = (lane & 16) >> 1;
    for (int t = wave; t < pairs; t += nwaves) {
        int tm = t / pairsN, tp = t - tm * pairsN;
        const uint16_t* Ap = A + (size_t)(tm * 16 + row) * K + off;
        const v16u* Bp = (const v16u*)Bs + (size_t)(tp * 2) * 32 + lane;
        v8f a0 = {}, a1 = {};
        for (int kb = 0; kb < K; kb += 32, Bp += (size_t)tilesN * 32) {
            union { v16u v; uint4 q[2]; } a;
            a.q[0] = *(const uint4*)(Ap + kb);
            a.q[1] = *(const uint4*)(Ap + kb + 16);
            v16u b0 = Bp[0];
            v16u b1 = Bp[32];
            a0 = wmma_bf16(a.v, b0, a0);
            a1 = wmma_bf16(a.v, b1, a1);
        }
        int m0 = tm * 16 + off;
        int n0 = tp * 32 + row;
        float r0[8], r1[8];
        float bv0 = bias ? bias[n0] : 0.f;
        float bv1 = bias ? bias[n0 + 16] : 0.f;
#pragma unroll
        for (int r = 0; r < 8; ++r) {
            float v0 = a0[r] + bv0, v1 = a1[r] + bv1;
            r0[r] = relu ? (v0 > 0.f ? v0 : 0.f) : v0;
            r1[r] = relu ? (v1 > 0.f ? v1 : 0.f) : v1;
        }
        if (Cf) {
            float* p = Cf + (size_t)m0 * N + n0;
#pragma unroll
            for (int r = 0; r < 8; ++r) { p[(size_t)r * N] = r0[r]; p[(size_t)r * N + 16] = r1[r]; }
        }
        if (Cb) {
            uint16_t* p = Cb + (size_t)m0 * N + n0;
#pragma unroll
            for (int r = 0; r < 8; ++r) { p[(size_t)r * N] = f2bf(r0[r]); p[(size_t)r * N + 16] = f2bf(r1[r]); }
        }
    }
}

// ---------------------------------------------------------------------------
// Fused NNConv, one graph per workgroup (8 wave32s).
//  - TN>=2 layers: wave = (mt, 1st N-tile), full K, TN interleaved accumulators
//  - TN==1 layers: wave = (k-half, mt, tile set {nb,nb+2}) -> 2 interleaved
//    accumulators, partials reduced through s_msg (store phase / add phase)
//  - root term via WMMA against pre-swizzled root (DIN>=64), else scalar
//  - fused scatter-add over tgt + bias + relu
// ---------------------------------------------------------------------------
template<int DIN, int DOUT>
__global__ __launch_bounds__(256)
void nnconv_kernel(const uint16_t* __restrict__ h_in,     // [4096, DIN]  bf16
                   const uint16_t* __restrict__ h2e,      // [8192, 64]   bf16
                   const uint16_t* __restrict__ w3sw,     // [KP, DOUT]   bf16 swizzled
                   const uint16_t* __restrict__ root_swz, // [DIN, DOUT]  bf16 swizzled (DIN>=64)
                   const float*    __restrict__ root,     // [DIN, DOUT]  f32 (DIN==4 path)
                   const float*    __restrict__ bias,     // [DOUT]
                   const int*      __restrict__ eidx,     // [2, 8192]
                   uint16_t*       __restrict__ h_out) {  // [4096, DOUT] bf16
    constexpr int LD  = (DIN == 4) ? 2 : (DIN == 64) ? 6 : (DIN == 128) ? 7 : 8;
    constexpr int KV  = 65 * DIN;
    constexpr int KP  = (KV + 31) & ~31;
    constexpr int NCH = KP / 32;
    constexpr int NT  = DOUT / 16;               // 4, 8, 16, 4
    constexpr int TN  = NT / 4;                  // 1, 2, 4, 1
    constexpr int T2  = (TN == 1) ? 2 : TN;      // tiles per wave

    const int g   = blockIdx.x;
    const int tid = threadIdx.x;

    __shared__ __align__(16) uint16_t s_hn[16 * DIN];
    __shared__ __align__(16) uint16_t s_hs[32 * DIN];
    __shared__ __align__(16) uint16_t s_h2[32 * 64];
    __shared__ float s_msg[32 * DOUT];
    __shared__ float s_root[(DIN >= 64) ? 16 * DOUT : 4];
    __shared__ int   s_src[32], s_tgt[32];

    if (tid < 32) {
        s_src[tid] = eidx[g * 32 + tid]        - g * 16;
        s_tgt[tid] = eidx[8192 + g * 32 + tid] - g * 16;
    }
    {
        const uint4* src = (const uint4*)(h_in + (size_t)g * 16 * DIN);
        uint4* dst = (uint4*)s_hn;
        for (int i = tid; i < 16 * DIN / 8; i += 256) dst[i] = src[i];
        const uint4* src2 = (const uint4*)(h2e + (size_t)g * 32 * 64);
        uint4* dst2 = (uint4*)s_h2;
        for (int i = tid; i < 32 * 64 / 8; i += 256) dst2[i] = src2[i];
    }
    __syncthreads();
    if constexpr (DIN >= 64) {
        constexpr int RV = DIN / 8;
        uint4* hs = (uint4*)s_hs;
        const uint4* hn = (const uint4*)s_hn;
        for (int i = tid; i < 32 * RV; i += 256) {
            int e = i / RV, f = i - e * RV;
            hs[i] = hn[s_src[e] * RV + f];
        }
    } else {
        for (int i = tid; i < 32 * DIN; i += 256) {
            int e = i >> LD, f = i & (DIN - 1);
            s_hs[i] = s_hn[s_src[e] * DIN + f];
        }
    }
    __syncthreads();

    const int wave = tid >> 5, lane = tid & 31;
    const int mt   = wave & 1;
    const int row  = lane & 15;
    const int off  = (lane & 16) >> 1;
    const int e    = mt * 16 + row;
    const v16u* Bt = (const v16u*)w3sw;

    int ntb[T2]; int c0, c1; bool addPhase = false;
    if constexpr (TN == 1) {
        int kh = wave >> 2;
        int nb = (wave >> 1) & 1;
        ntb[0] = nb; ntb[1] = nb + 2;
        c0 = kh ? NCH / 2 : 0;
        c1 = kh ? NCH : NCH / 2;
        addPhase = (kh == 1);
    } else {
        int w2 = wave >> 1;
#pragma unroll
        for (int t = 0; t < T2; ++t) ntb[t] = w2 + t * 4;
        c0 = 0; c1 = NCH;
    }

    v8f acc[T2];
#pragma unroll
    for (int t = 0; t < T2; ++t) acc[t] = (v8f){};

    for (int ch = c0; ch < c1; ++ch) {
        int kb = ch << 5;
        union { v16u v; uint4 q[2]; v8bf h[2]; uint16_t s[16]; } a;
        if constexpr (DIN >= 64) {
            // h2 coefficient constant across each 8-wide K run -> packed bf16 mul
#pragma unroll
            for (int half = 0; half < 2; ++half) {
                int kk0 = kb + (half << 4) + off;
                int k = kk0 >> LD;
                __bf16 cb = (k == 64) ? (__bf16)1.0f
                                      : *(const __bf16*)&s_h2[e * 64 + k];
                v8bf hv = *(const v8bf*)&s_hs[e * DIN + (kk0 & (DIN - 1))];
                a.h[half] = hv * cb;
            }
        } else {   // DIN == 4: K runs cross k blocks; scalar build (KP tiny)
#pragma unroll
            for (int j = 0; j < 16; ++j) {
                int kk = kb + kmap(j, lane);
                int k = kk >> LD, f = kk & (DIN - 1);
                float av;
                if (kk >= KV)     av = 0.f;
                else if (k == 64) av = bf2f(s_hs[e * DIN + f]);
                else              av = bf2f(s_h2[e * 64 + k]) * bf2f(s_hs[e * DIN + f]);
                a.s[j] = f2bf(av);
            }
        }
        const v16u* Bp = Bt + ((size_t)ch * NT) * 32 + lane;
        if (ch + 1 < c1)
            __builtin_prefetch(Bt + ((size_t)(ch + 1) * NT + ntb[0]) * 32 + lane, 0, 1);
#pragma unroll
        for (int t = 0; t < T2; ++t)
            acc[t] = wmma_bf16(a.v, Bp[ntb[t] * 32], acc[t]);
    }

    if (!addPhase) {
#pragma unroll
        for (int t = 0; t < T2; ++t)
#pragma unroll
            for (int r = 0; r < 8; ++r)
                s_msg[(mt * 16 + r + off) * DOUT + ntb[t] * 16 + row] = acc[t][r];
    }

    // root term: 16xDINxDOUT WMMA GEMM per graph (DIN>=64)
    if constexpr (DIN >= 64) {
        const v16u* Rt = (const v16u*)root_swz;
        for (int nt = wave; nt < NT; nt += 8) {
            v8f rc = {};
#pragma unroll
            for (int ch2 = 0; ch2 < DIN / 32; ++ch2) {
                union { v16u v; uint4 q[2]; } ar;
                const uint16_t* hp = &s_hn[row * DIN + (ch2 << 5) + off];
                ar.q[0] = *(const uint4*)hp;
                ar.q[1] = *(const uint4*)(hp + 16);
                rc = wmma_bf16(ar.v, Rt[((size_t)ch2 * NT + nt) * 32 + lane], rc);
            }
#pragma unroll
            for (int r = 0; r < 8; ++r)
                s_root[(r + off) * DOUT + nt * 16 + row] = rc[r];
        }
    }
    __syncthreads();
    if (addPhase) {
#pragma unroll
        for (int t = 0; t < T2; ++t)
#pragma unroll
            for (int r = 0; r < 8; ++r)
                s_msg[(mt * 16 + r + off) * DOUT + ntb[t] * 16 + row] += acc[t][r];
    }
    __syncthreads();

    // Fused: segment-sum over tgt + root + bias + relu
    for (int idx = tid; idx < 16 * DOUT; idx += 256) {
        int n = idx / DOUT, o = idx - n * DOUT;
        float acc2 = bias[o];
        if constexpr (DIN >= 64) {
            acc2 += s_root[idx];
        } else {
            for (int i = 0; i < DIN; ++i)
                acc2 += bf2f(s_hn[n * DIN + i]) * root[(size_t)i * DOUT + o];
        }
        for (int ee = 0; ee < 32; ++ee)
            if (s_tgt[ee] == n) acc2 += s_msg[ee * DOUT + o];
        acc2 = acc2 > 0.f ? acc2 : 0.f;
        h_out[((size_t)g * 16 + n) * DOUT + o] = f2bf(acc2);
    }
}

// ---------------------------------------------------------------------------
// Readout: out[b] = concat(h[in_b], h[out_b], h[gnd_b]) @ out_w + out_b
// ---------------------------------------------------------------------------
__global__ void final_out_kernel(const float* __restrict__ t3,
                                 const int* __restrict__ iin, const int* __restrict__ iout,
                                 const int* __restrict__ ignd,
                                 const float* __restrict__ ow, const float* __restrict__ ob,
                                 float* __restrict__ out) {
    int b = blockIdx.x * blockDim.x + threadIdx.x;
    if (b >= 256) return;
    float acc = ob[0];
    const float* r0 = t3 + (size_t)iin[b]  * 64;
    const float* r1 = t3 + (size_t)iout[b] * 64;
    const float* r2 = t3 + (size_t)ignd[b] * 64;
    for (int j = 0; j < 64; ++j) acc += r0[j] * ow[j];
    for (int j = 0; j < 64; ++j) acc += r1[j] * ow[64 + j];
    for (int j = 0; j < 64; ++j) acc += r2[j] * ow[128 + j];
    out[b] = acc;
}

// ---------------------------------------------------------------------------
// Host orchestration
// ---------------------------------------------------------------------------
extern "C" void kernel_launch(void* const* d_in, const int* in_sizes, int n_in,
                              void* d_out, int out_size, void* d_ws, size_t ws_size,
                              hipStream_t stream) {
    (void)in_sizes; (void)n_in; (void)out_size; (void)ws_size;

    const float* x    = (const float*)d_in[0];
    const int*   eidx = (const int*)  d_in[1];
    const float* ea1  = (const float*)d_in[2];
    const float* ea2  = (const float*)d_in[3];
    const int*   iin  = (const int*)  d_in[4];
    const int*   iou  = (const int*)  d_in[5];
    const int*   ignd = (const int*)  d_in[6];
    auto CP = [&](int l, int k) { return (const float*)d_in[7 + 8 * l + k]; };
    const float* lin1_w = (const float*)d_in[39]; const float* lin1_b = (const float*)d_in[40];
    const float* lin2_w = (const float*)d_in[41]; const float* lin2_b = (const float*)d_in[42];
    const float* lin3_w = (const float*)d_in[43]; const float* lin3_b = (const float*)d_in[44];
    const float* out_w  = (const float*)d_in[45]; const float* out_b  = (const float*)d_in[46];

    const int DINS[4]  = {4, 64, 128, 256};
    const int DOUTS[4] = {64, 128, 256, 64};
    int KP[4];
    for (int l = 0; l < 4; ++l) KP[l] = (65 * DINS[l] + 31) & ~31;

    char* ws = (char*)d_ws;
    size_t off = 0;
    auto alloc = [&](size_t bytes) { char* p = ws + off; off = (off + bytes + 255) & ~(size_t)255; return p; };
    float*    ea   = (float*)   alloc(8192 * 3 * 4);
    uint16_t* h0   = (uint16_t*)alloc(4096 * 4 * 2);
    uint16_t* hb[4];
    for (int l = 0; l < 4; ++l) hb[l] = (uint16_t*)alloc((size_t)4096 * DOUTS[l] * 2);
    uint16_t* h1t  = (uint16_t*)alloc((size_t)8192 * 64 * 2);
    uint16_t* h2t  = (uint16_t*)alloc((size_t)8192 * 64 * 2);
    uint16_t* w2bf[4]; uint16_t* w3c[4]; uint16_t* rsw[4];
    for (int l = 0; l < 4; ++l) w2bf[l] = (uint16_t*)alloc(64 * 64 * 2);
    for (int l = 0; l < 4; ++l) w3c[l]  = (uint16_t*)alloc((size_t)KP[l] * DOUTS[l] * 2);
    rsw[0] = nullptr;
    for (int l = 1; l < 4; ++l) rsw[l] = (uint16_t*)alloc((size_t)DINS[l] * DOUTS[l] * 2);
    uint16_t* l1bf = (uint16_t*)alloc(64 * 128 * 2);
    uint16_t* l2bf = (uint16_t*)alloc(128 * 64 * 2);
    uint16_t* l3bf = (uint16_t*)alloc(64 * 64 * 2);
    uint16_t* t1   = (uint16_t*)alloc((size_t)4096 * 128 * 2);
    uint16_t* t2   = (uint16_t*)alloc((size_t)4096 * 64 * 2);
    float*    t3   = (float*)   alloc((size_t)4096 * 64 * 4);

    auto blocks = [](int n) { return (n + 255) / 256; };
    auto swz = [&](const float* s, uint16_t* d, int K, int N) {
        swz_w_kernel<<<blocks(K * N), 256, 0, stream>>>(s, d, K, N);
    };
    auto gemm = [&](const uint16_t* A, const uint16_t* B, const float* bias,
                    uint16_t* Cb, float* Cf, int M, int N, int K, int relu) {
        int pairs = (M / 16) * (N / 32);
        wmma_gemm_kernel<<<(pairs + 7) / 8, 256, 0, stream>>>(A, B, bias, Cb, Cf, M, N, K, relu);
    };

    // ---- prep ----
    ea_sum_kernel<<<blocks(8192 * 3), 256, 0, stream>>>(ea1, ea2, ea, 8192 * 3);
    cvt_bf16_kernel<<<blocks(4096 * 4), 256, 0, stream>>>(x, h0, 4096 * 4);
    for (int l = 0; l < 4; ++l) {
        swz(CP(l, 2), w2bf[l], 64, 64);
        int n = KP[l] * DOUTS[l];
        w3cat_swz_kernel<<<blocks(n), 256, 0, stream>>>(CP(l, 4), CP(l, 5), w3c[l],
                                                        DINS[l], DOUTS[l], KP[l]);
        if (l >= 1) swz(CP(l, 6), rsw[l], DINS[l], DOUTS[l]);
    }
    swz(lin1_w, l1bf, 64, 128);
    swz(lin2_w, l2bf, 128, 64);
    swz(lin3_w, l3bf, 64, 64);

    // ---- four fused NNConv layers ----
    for (int l = 0; l < 4; ++l) {
        edge_h1_kernel<<<blocks(8192 * 64), 256, 0, stream>>>(ea, CP(l, 0), CP(l, 1), h1t);
        gemm(h1t, w2bf[l], CP(l, 3), h2t, nullptr, 8192, 64, 64, 1);
        const uint16_t* hin = (l == 0) ? h0 : hb[l - 1];
        switch (l) {
        case 0: nnconv_kernel<4, 64>  <<<256, 256, 0, stream>>>(hin, h2t, w3c[0], nullptr, CP(0,6), CP(0,7), eidx, hb[0]); break;
        case 1: nnconv_kernel<64, 128><<<256, 256, 0, stream>>>(hin, h2t, w3c[1], rsw[1], CP(1,6), CP(1,7), eidx, hb[1]); break;
        case 2: nnconv_kernel<128,256><<<256, 256, 0, stream>>>(hin, h2t, w3c[2], rsw[2], CP(2,6), CP(2,7), eidx, hb[2]); break;
        case 3: nnconv_kernel<256, 64><<<256, 256, 0, stream>>>(hin, h2t, w3c[3], rsw[3], CP(3,6), CP(3,7), eidx, hb[3]); break;
        }
    }

    // ---- linear chain (no activations) + readout ----
    gemm(hb[3], l1bf, lin1_b, t1, nullptr, 4096, 128, 64, 0);
    gemm(t1,    l2bf, lin2_b, t2, nullptr, 4096, 64, 128, 0);
    gemm(t2,    l3bf, lin3_b, nullptr, t3, 4096, 64, 64, 0);
    final_out_kernel<<<1, 256, 0, stream>>>(t3, iin, iou, ignd, out_w, out_b, (float*)d_out);
}
